// DimeNet_48455821033951
// MI455X (gfx1250) — matrix operand order repaired and checked by reading, not compiled, not run on hardware
//
#include <hip/hip_runtime.h>
#include <math.h>

// ---------------------------------------------------------------- constants
static constexpr int kN = 50000, kE = 200000, kT = 200000, kB = 128;

typedef __attribute__((ext_vector_type(16))) _Float16 v16h;
typedef __attribute__((ext_vector_type(8)))  float    v8f;
typedef __attribute__((ext_vector_type(4)))  unsigned int u32x4;
typedef __attribute__((ext_vector_type(8)))  int    i32x8;
typedef __attribute__((ext_vector_type(4)))  int    i32x4;

struct U32x8 { uint4 a, b; };   // 32 bytes == v16h

__device__ __forceinline__ float swishf(float x) { return x / (1.f + __expf(-x)); }

// ---------------------------------------------------------------- WMMA GEMM
// out[M,128] = epilogue( A[M,K](f16, optionally row-gathered) @ B[K,128] )
// BT is f16 column-major: BT[n*K + k], K padded to multiple of 32, K <= 128.
// Block: 256 threads / 8 waves; 64 rows x 128 cols. Wave w owns col-tile w
// and 4 row-tiles with independent accumulators (B frag reused 4x).
struct GemmP {
  const _Float16* A; const _Float16* BT; const float* bias;
  const int* gather;        // row index for A (and mul) or null
  const float* mul;         // [*,128] gate, row-indexed like A
  const float* addv;        // [M,128] residual add, row-indexed by out row
  float* out32; _Float16* out16;
  int M, K;
  int pre_act;              // swish before mul/add
  int post_act;             // swish after add
  int sw16;                 // out16 = f16(swish(val)) instead of f16(val)
};

__global__ __launch_bounds__(256) void k_gemm(GemmP g) {
  __shared__ _Float16 sA[64 * 128];            // 16 KB, K <= 128
  const int row0 = blockIdx.x * 64;
  const int tid  = threadIdx.x;
  const int K    = g.K;

  bool manual = true;
#if __has_builtin(__builtin_amdgcn_tensor_load_to_lds)
  if (g.gather == nullptr && row0 + 64 <= g.M) {
    manual = false;
    if (tid < 32) {   // one wave drives the Tensor Data Mover
      const unsigned long long ga =
          (unsigned long long)(const void*)(g.A + (size_t)row0 * K);
      const unsigned lds = (unsigned)(size_t)(&sA[0]);
      const unsigned td0 = (unsigned)K;              // elements per row
      const unsigned td1 = (unsigned)(g.M - row0);   // rows available
      u32x4 g0;
      g0[0] = 1u;                                    // count=1, user descriptor
      g0[1] = lds;                                   // D#.lds_addr (bytes)
      g0[2] = (unsigned)(ga & 0xffffffffu);          // global_addr[31:0]
      g0[3] = (unsigned)((ga >> 32) & 0x01ffffffu)   // global_addr[56:32]
              | (2u << 30);                          // type = 2 ("image")
      i32x8 g1;
      g1[0] = (int)(1u << 16);                       // data_size=1 (2 bytes)
      g1[1] = (int)((td0 & 0xffffu) << 16);          // tensor_dim0[15:0]
      g1[2] = (int)((td0 >> 16) | ((td1 & 0xffffu) << 16)); // td0[31:16], td1[15:0]
      g1[3] = (int)(((td1 >> 16) & 0xffffu) | ((td0 & 0xffffu) << 16)); // td1 hi, tile_dim0=K
      g1[4] = (int)64;                               // tile_dim1=64, tile_dim2=0
      g1[5] = (int)K;                                // tensor_dim0_stride[31:0]
      g1[6] = 0; g1[7] = 0;
      i32x4 z4 = {0, 0, 0, 0};
#if __clang_major__ >= 23
      i32x8 z8 = {0, 0, 0, 0, 0, 0, 0, 0};
      __builtin_amdgcn_tensor_load_to_lds(g0, g1, z4, z4, z8, 0);
#else
      __builtin_amdgcn_tensor_load_to_lds(g0, g1, z4, z4, 0);
#endif
      __builtin_amdgcn_s_wait_tensorcnt(0);
    }
  }
#endif
  if (manual) {
    // cooperative A-tile load (zero-fill rows past M), supports row gather
    for (int idx = tid * 8; idx < 64 * K; idx += 256 * 8) {
      const int r = idx / K;
      const int grow = row0 + r;
      uint4 val = make_uint4(0u, 0u, 0u, 0u);
      if (grow < g.M) {
        const int arow = g.gather ? g.gather[grow] : grow;
        val = *(const uint4*)(g.A + (size_t)arow * K + (idx - r * K));
      }
      *(uint4*)(sA + idx) = val;
    }
  }
  __syncthreads();

  const int wave = tid >> 5, lane = tid & 31;
  const int n0 = wave * 16;
  const int mlow = lane & 15, hi = lane >> 4;
  v8f acc[4] = {};
  const _Float16* Bp = g.BT + (size_t)(n0 + mlow) * K + hi * 8;
  const int aoff = mlow * K + hi * 8;
  for (int kb = 0; kb < K; kb += 32) {
    U32x8 tb;
    tb.a = *(const uint4*)(Bp + kb);
    tb.b = *(const uint4*)(Bp + kb + 16);
    const v16h bv = __builtin_bit_cast(v16h, tb);
#pragma unroll
    for (int rt = 0; rt < 4; ++rt) {
      const _Float16* Ap = sA + rt * 16 * K + aoff;
      U32x8 ta;
      ta.a = *(const uint4*)(Ap + kb);
      ta.b = *(const uint4*)(Ap + kb + 16);
      const v16h av = __builtin_bit_cast(v16h, ta);
      acc[rt] = __builtin_amdgcn_wmma_f32_16x16x32_f16(false, av, false, bv,
                                                       (short)0, acc[rt],
                                                       false, false);
    }
  }
  const int n = n0 + mlow;
  const float bn = g.bias ? g.bias[n] : 0.f;
#pragma unroll
  for (int rt = 0; rt < 4; ++rt) {
    for (int v = 0; v < 8; ++v) {
      const int grow = row0 + rt * 16 + hi * 8 + v;
      if (grow >= g.M) continue;
      const size_t o = (size_t)grow * 128 + n;
      float t = acc[rt][v] + bn;
      if (g.pre_act) t = swishf(t);
      if (g.mul) {
        const int arow = g.gather ? g.gather[grow] : grow;
        t *= g.mul[(size_t)arow * 128 + n];
      }
      if (g.addv) t += g.addv[o];
      if (g.post_act) t = swishf(t);
      if (g.out32) g.out32[o] = t;
      if (g.out16) g.out16[o] = (_Float16)(g.sw16 ? swishf(t) : t);
    }
  }
}

// ------------------------------------------------- fused bilinear (K=1024)
// tm[t,k] = sum_{h,b} xkj[t,h]*sbfp[t,b]*W[h,b,k]  ==  y[t,1024] @ Wr[1024,128]
struct BilP {
  const _Float16* xkj;   // [T,128] f16
  const float* sbfp;     // [T,8]
  const _Float16* BT;    // [128,1024] f16 (col-major of Wr)
  float* out32;          // [T,128]
  int M;
};
__global__ __launch_bounds__(256) void k_bilinear(BilP g) {
  __shared__ _Float16 sY[16 * 1024];           // 32 KB
  __shared__ float sS[16 * 8];
  const int row0 = blockIdx.x * 16;
  const int tid = threadIdx.x;
  if (tid < 128) {
    const int r = tid >> 3, b = tid & 7, grow = row0 + r;
    sS[tid] = (grow < g.M) ? g.sbfp[(size_t)grow * 8 + b] : 0.f;
  }
  __syncthreads();
  {
    const int r = tid >> 4, h0 = (tid & 15) * 8, grow = row0 + r;
    for (int hh = 0; hh < 8; ++hh) {
      const int h = h0 + hh;
      const float xv = (grow < g.M) ? (float)g.xkj[(size_t)grow * 128 + h] : 0.f;
      for (int b = 0; b < 8; ++b)
        sY[r * 1024 + h * 8 + b] = (_Float16)(xv * sS[r * 8 + b]);
    }
  }
  __syncthreads();
  const int wave = tid >> 5, lane = tid & 31;
  const int n0 = wave * 16, mlow = lane & 15, hi = lane >> 4;
  v8f acc0 = {}, acc1 = {};                    // two chains for XDL overlap
  const _Float16* Bp = g.BT + (size_t)(n0 + mlow) * 1024 + hi * 8;
  const _Float16* Ap = sY + mlow * 1024 + hi * 8;
  for (int kb = 0; kb < 1024; kb += 64) {
    U32x8 ta, tb;
    ta.a = *(const uint4*)(Ap + kb);
    ta.b = *(const uint4*)(Ap + kb + 16);
    tb.a = *(const uint4*)(Bp + kb);
    tb.b = *(const uint4*)(Bp + kb + 16);
    acc0 = __builtin_amdgcn_wmma_f32_16x16x32_f16(false, __builtin_bit_cast(v16h, ta),
                                                  false, __builtin_bit_cast(v16h, tb),
                                                  (short)0, acc0, false, false);
    ta.a = *(const uint4*)(Ap + kb + 32);
    ta.b = *(const uint4*)(Ap + kb + 48);
    tb.a = *(const uint4*)(Bp + kb + 32);
    tb.b = *(const uint4*)(Bp + kb + 48);
    acc1 = __builtin_amdgcn_wmma_f32_16x16x32_f16(false, __builtin_bit_cast(v16h, ta),
                                                  false, __builtin_bit_cast(v16h, tb),
                                                  (short)0, acc1, false, false);
  }
  for (int v = 0; v < 8; ++v) {
    const int grow = row0 + hi * 8 + v;
    if (grow < g.M)
      g.out32[(size_t)grow * 128 + n0 + mlow] = acc0[v] + acc1[v];
  }
}

// ---------------------------------------------------------------- utilities
__global__ void k_wt(const float* src, _Float16* dst, int K, int Kpad, int Nc, int rowOff) {
  const int i = blockIdx.x * 256 + threadIdx.x;
  if (i >= Nc * Kpad) return;
  const int n = i / Kpad, r = i - n * Kpad;
  dst[i] = (r < K) ? (_Float16)src[(size_t)(rowOff + r) * Nc + n] : (_Float16)0.f;
}
__global__ void k_cast16(const float* s, _Float16* d, size_t n, int sw) {
  const size_t i = (size_t)blockIdx.x * 256 + threadIdx.x;
  if (i < n) { float t = s[i]; if (sw) t = swishf(t); d[i] = (_Float16)t; }
}
__global__ void k_mul16(const float* a, const float* b, _Float16* d, size_t n) {
  const size_t i = (size_t)blockIdx.x * 256 + threadIdx.x;
  if (i < n) d[i] = (_Float16)(a[i] * b[i]);
}
__global__ void k_addsw16(const float* a, const float* b, float* v, _Float16* d, size_t n) {
  const size_t i = (size_t)blockIdx.x * 256 + threadIdx.x;
  if (i < n) { const float t = a[i] + b[i]; v[i] = t; d[i] = (_Float16)swishf(t); }
}
__global__ void k_zero(float* p, size_t n) {
  const size_t i = (size_t)blockIdx.x * 256 + threadIdx.x;
  if (i < n) p[i] = 0.f;
}
__global__ void k_scatter(const float* src, const int* idx, float* dst, int rows) {
  const int r = blockIdx.x, c = threadIdx.x;   // blockDim == 128
  if (r < rows) atomicAdd(&dst[(size_t)idx[r] * 128 + c], src[(size_t)r * 128 + c]);
}
__global__ void k_rbf(const float* ea, float* rbf, _Float16* r16, int E) {
  const int e = blockIdx.x * 256 + threadIdx.x;
  if (e >= E) return;
  const float* p = ea + (size_t)e * 16;
  float s = 0.f;
  for (int i = 0; i < 16; ++i) s += p[i] * p[i];
  const float dist = sqrtf(s);
  const float inv = 1.f / (dist + 1e-8f);
  const float w0 = 3.14159265358979323846f / 5.f;
  for (int f = 0; f < 16; ++f) {
    const float v = sinf((f + 1) * w0 * dist) * inv;
    rbf[(size_t)e * 16 + f] = v;
    r16[(size_t)e * 32 + f] = (_Float16)v;
  }
  for (int f = 16; f < 32; ++f) r16[(size_t)e * 32 + f] = (_Float16)0.f;
}
__global__ void k_ea16(const float* ea, _Float16* d, int E) {
  const int i = blockIdx.x * 256 + threadIdx.x;
  if (i >= E * 32) return;
  const int e = i >> 5, c = i & 31;
  d[i] = (c < 16) ? (_Float16)ea[(size_t)e * 16 + c] : (_Float16)0.f;
}
__global__ void k_sbf(const float* ea, const int* kj, const int* ji, float* sbf, int T) {
  const int t = blockIdx.x * 256 + threadIdx.x;
  if (t >= T) return;
  const float* a = ea + (size_t)kj[t] * 16;
  const float* b = ea + (size_t)ji[t] * 16;
  float na = 0.f, nb = 0.f, dp = 0.f;
  for (int i = 0; i < 16; ++i) { na += a[i]*a[i]; nb += b[i]*b[i]; dp += a[i]*b[i]; }
  na = fmaxf(sqrtf(na), 1e-8f); nb = fmaxf(sqrtf(nb), 1e-8f);
  float c = dp / (na * nb);
  c = fminf(0.999f, fmaxf(-0.999f, c));
  const float ang = acosf(c);
  for (int s = 0; s < 7; ++s) sbf[(size_t)t * 8 + s] = sinf((s + 1) * ang);
  sbf[(size_t)t * 8 + 7] = 0.f;
}
__global__ void k_sbfproj(const float* sbf, const float* w, float* out, int T) {
  const int t = blockIdx.x * 256 + threadIdx.x;
  if (t >= T) return;
  const float* s = sbf + (size_t)t * 8;
  for (int b = 0; b < 8; ++b) {
    float acc = 0.f;
    for (int k = 0; k < 7; ++k) acc += s[k] * w[k * 8 + b];
    out[(size_t)t * 8 + b] = acc;
  }
}
__global__ __launch_bounds__(256) void k_head(
    const float* hg, const float* ph, const float* te, const float* rf,
    const int* box, const float* bemb,
    const float* c0w, const float* c0b, const float* c1w, const float* c1b,
    const float* m0w, const float* m0b, const float* m1w, const float* m1b,
    const float* m2w, const float* m2b, float* out) {
  __shared__ float v160[160], o256[256], o128[128], c64[64], cin[11];
  const int b = blockIdx.x, tid = threadIdx.x;
  if (tid == 0) { cin[0] = ph[b]; cin[1] = te[b]; cin[2] = rf[b]; }
  if (tid < 8) cin[3 + tid] = bemb[box[b] * 8 + tid];
  if (tid < 128) v160[tid] = hg[(size_t)b * 128 + tid];
  __syncthreads();
  if (tid < 64) {
    float s = c0b[tid];
    for (int k = 0; k < 11; ++k) s += cin[k] * c0w[k * 64 + tid];
    c64[tid] = fmaxf(s, 0.f);
  }
  __syncthreads();
  if (tid < 32) {
    float s = c1b[tid];
    for (int k = 0; k < 64; ++k) s += c64[k] * c1w[k * 32 + tid];
    v160[128 + tid] = s;
  }
  __syncthreads();
  {
    float s = m0b[tid];
    for (int k = 0; k < 160; ++k) s += v160[k] * m0w[k * 256 + tid];
    o256[tid] = fmaxf(s, 0.f);
  }
  __syncthreads();
  if (tid < 128) {
    float s = m1b[tid];
    for (int k = 0; k < 256; ++k) s += o256[k] * m1w[k * 128 + tid];
    o128[tid] = fmaxf(s, 0.f);
  }
  __syncthreads();
  if (tid == 0) {
    float s = m2b[0];
    for (int k = 0; k < 128; ++k) s += o128[k] * m2w[k];
    out[b] = s;
  }
}

// ---------------------------------------------------------------- driver
extern "C" void kernel_launch(void* const* d_in, const int* in_sizes, int n_in,
                              void* d_out, int out_size, void* d_ws, size_t ws_size,
                              hipStream_t stream) {
  (void)in_sizes; (void)n_in; (void)out_size; (void)ws_size;
  const float* x     = (const float*)d_in[0];
  const int*   ei    = (const int*)d_in[1];
  const float* ea    = (const float*)d_in[2];
  const int*   batch = (const int*)d_in[3];
  const int*   ti    = (const int*)d_in[4];
  const float* ph    = (const float*)d_in[5];
  const float* te    = (const float*)d_in[6];
  const float* rf    = (const float*)d_in[7];
  const int*   box   = (const int*)d_in[8];
  auto P = [&](int i) { return (const float*)d_in[i]; };
  const int* gi = ei;      const int* gj = ei + kE;
  const int* kj = ti;      const int* ji = ti + kT;

  // -------- workspace layout
  char* wsb = (char*)d_ws; size_t woff = 0;
  auto alloc = [&](size_t bytes) -> void* {
    woff = (woff + 255) & ~(size_t)255;
    void* p = wsb + woff; woff += bytes; return p;
  };
  float*     rbf32 = (float*)alloc((size_t)kE * 16 * 4);
  _Float16*  rbf16 = (_Float16*)alloc((size_t)kE * 32 * 2);
  _Float16*  ea16  = (_Float16*)alloc((size_t)kE * 32 * 2);
  _Float16*  x16   = (_Float16*)alloc((size_t)kN * 64 * 2);
  float*     sbf32 = (float*)alloc((size_t)kT * 8 * 4);
  float*     sbfp  = (float*)alloc((size_t)kT * 8 * 4);
  float*     m32   = (float*)alloc((size_t)kE * 128 * 4);
  float*     t1    = (float*)alloc((size_t)kE * 128 * 4);
  float*     t2    = (float*)alloc((size_t)kE * 128 * 4);
  float*     aggr  = (float*)alloc((size_t)kE * 128 * 4);
  float*     tm32  = (float*)alloc((size_t)kT * 128 * 4);
  float*     ge32  = (float*)alloc((size_t)kE * 128 * 4);
  _Float16*  m16   = (_Float16*)alloc((size_t)kE * 128 * 2);
  _Float16*  a16   = (_Float16*)alloc((size_t)kE * 128 * 2);
  _Float16*  b16   = (_Float16*)alloc((size_t)kE * 128 * 2);
  _Float16*  c16   = (_Float16*)alloc((size_t)kE * 128 * 2);
  _Float16*  xkj16 = (_Float16*)alloc((size_t)kT * 128 * 2);
  float*     h32   = (float*)alloc((size_t)kN * 128 * 4);
  float*     hg    = (float*)alloc((size_t)kB * 128 * 4);

  _Float16 *wt_node = (_Float16*)alloc(128*64*2);
  _Float16 *wt_edge = (_Float16*)alloc(128*32*2);
  _Float16 *wt_rbf  = (_Float16*)alloc(128*32*2);
  _Float16 *wt_lin[3];
  for (int p = 0; p < 3; ++p) wt_lin[p] = (_Float16*)alloc(128*128*2);
  _Float16 *wt_ji[4], *wt_kj[4], *wt_out[4], *wt_lrbf[4], *wt_bil[4], *wt_ir[4][4];
  for (int l = 0; l < 4; ++l) {
    wt_ji[l]   = (_Float16*)alloc(128*128*2);
    wt_kj[l]   = (_Float16*)alloc(128*128*2);
    wt_out[l]  = (_Float16*)alloc(128*128*2);
    wt_lrbf[l] = (_Float16*)alloc(128*32*2);
    wt_bil[l]  = (_Float16*)alloc((size_t)128*1024*2);
    for (int r = 0; r < 4; ++r) wt_ir[l][r] = (_Float16*)alloc(128*128*2);
  }
  _Float16 *wt_ob[5][7];
  for (int o = 0; o < 5; ++o)
    for (int r = 0; r < 7; ++r) wt_ob[o][r] = (_Float16*)alloc(128*128*2);

  // -------- helpers
  auto conv = [&](const float* src, _Float16* dst, int K, int Kpad, int Nc, int rowOff) {
    const int n = Nc * Kpad;
    k_wt<<<(n + 255) / 256, 256, 0, stream>>>(src, dst, K, Kpad, Nc, rowOff);
  };
  auto gemm = [&](const _Float16* A, const _Float16* BT, const float* bias,
                  const int* gather, const float* mul, const float* addv,
                  float* out32, _Float16* out16, int M, int K,
                  int pre, int post, int sw16) {
    GemmP g{A, BT, bias, gather, mul, addv, out32, out16, M, K, pre, post, sw16};
    k_gemm<<<(M + 63) / 64, 256, 0, stream>>>(g);
  };
  auto g1d = [](size_t n) { return (unsigned)((n + 255) / 256); };
  const size_t EH = (size_t)kE * 128;

  // -------- weight conversion (param leaves: sorted-dict pytree order, base 9)
  conv(P(19), wt_node, 64, 64, 128, 0);
  conv(P(15), wt_edge, 16, 32, 128, 0);
  conv(P(21), wt_rbf,  16, 32, 128, 0);
  for (int p = 0; p < 3; ++p) conv(P(17), wt_lin[p], 128, 128, 128, p * 128);
  for (int l = 0; l < 4; ++l) {
    const int L = 22 + 17 * l;
    conv(P(L + 0), wt_bil[l], 1024, 1024, 128, 0);
    conv(P(L + 2), wt_ji[l],  128, 128, 128, 0);
    conv(P(L + 4), wt_kj[l],  128, 128, 128, 0);
    conv(P(L + 6), wt_out[l], 128, 128, 128, 0);
    conv(P(L + 7), wt_lrbf[l], 16, 32, 128, 0);
    conv(P(L + 10), wt_ir[l][0], 128, 128, 128, 0);
    conv(P(L + 12), wt_ir[l][1], 128, 128, 128, 0);
    conv(P(L + 14), wt_ir[l][2], 128, 128, 128, 0);
    conv(P(L + 16), wt_ir[l][3], 128, 128, 128, 0);
  }
  for (int o = 0; o < 5; ++o) {
    const int O = 96 + 14 * o;
    conv(P(O + 1), wt_ob[o][0], 128, 128, 128, 0);
    for (int r = 0; r < 3; ++r) {
      conv(P(O + 3 + 4 * r), wt_ob[o][1 + 2 * r], 128, 128, 128, 0);
      conv(P(O + 5 + 4 * r), wt_ob[o][2 + 2 * r], 128, 128, 128, 0);
    }
  }

  // -------- bases
  k_rbf<<<(kE + 255) / 256, 256, 0, stream>>>(ea, rbf32, rbf16, kE);
  k_ea16<<<g1d((size_t)kE * 32), 256, 0, stream>>>(ea, ea16, kE);
  k_cast16<<<g1d((size_t)kN * 64), 256, 0, stream>>>(x, x16, (size_t)kN * 64, 0);
  k_sbf<<<(kT + 255) / 256, 256, 0, stream>>>(ea, kj, ji, sbf32, kT);

  // -------- embedding: m = swish(cat(xi*xj, e, r) @ Wlin + b)
  gemm(x16, wt_node, P(18), gi, nullptr, nullptr, t1, nullptr, kE, 64, 0, 0, 0);
  gemm(x16, wt_node, P(18), gj, nullptr, nullptr, t2, nullptr, kE, 64, 0, 0, 0);
  k_mul16<<<g1d(EH), 256, 0, stream>>>(t1, t2, a16, EH);
  gemm(ea16,  wt_edge, P(14), nullptr, nullptr, nullptr, nullptr, b16, kE, 32, 0, 0, 0);
  gemm(rbf16, wt_rbf,  P(20), nullptr, nullptr, nullptr, nullptr, c16, kE, 32, 0, 0, 0);
  gemm(a16, wt_lin[0], nullptr, nullptr, nullptr, nullptr, t1, nullptr, kE, 128, 0, 0, 0);
  gemm(b16, wt_lin[1], nullptr, nullptr, nullptr, t1,      t1, nullptr, kE, 128, 0, 0, 0);
  gemm(c16, wt_lin[2], P(16),  nullptr, nullptr, t1,      m32, m16,    kE, 128, 0, 1, 0);

  // -------- out blocks accumulate into h
  k_zero<<<g1d((size_t)kN * 128), 256, 0, stream>>>(h32, (size_t)kN * 128);
  auto run_out_block = [&](int o) {
    const int O = 96 + 14 * o;
    k_cast16<<<g1d(EH), 256, 0, stream>>>(m32, a16, EH, 1);
    const float* cur = m32;
    for (int r = 0; r < 3; ++r) {
      gemm(a16, wt_ob[o][1 + 2 * r], P(O + 2 + 4 * r), nullptr, nullptr, nullptr,
           nullptr, b16, kE, 128, 0, 1, 0);
      gemm(b16, wt_ob[o][2 + 2 * r], P(O + 4 + 4 * r), nullptr, nullptr, cur,
           t1, a16, kE, 128, 0, 0, 1);
      cur = t1;
    }
    k_cast16<<<g1d(EH), 256, 0, stream>>>(t1, a16, EH, 0);
    gemm(a16, wt_ob[o][0], P(O + 0), nullptr, nullptr, nullptr, t2, nullptr,
         kE, 128, 0, 1, 0);
    k_scatter<<<kE, 128, 0, stream>>>(t2, gj, h32, kE);
  };
  run_out_block(0);

  // -------- interaction layers
  for (int l = 0; l < 4; ++l) {
    const int L = 22 + 17 * l;
    gemm(rbf16, wt_lrbf[l], nullptr, nullptr, nullptr, nullptr, ge32, nullptr, kE, 32, 0, 0, 0);
    k_sbfproj<<<(kT + 255) / 256, 256, 0, stream>>>(sbf32, P(L + 8), sbfp, kT);
    // x_ji = swish(lin_ji(m))
    gemm(m16, wt_ji[l], P(L + 1), nullptr, nullptr, nullptr, t1, nullptr, kE, 128, 0, 1, 0);
    // x_kj = swish(lin_kj(m[kj])) * gate[kj]
    gemm(m16, wt_kj[l], P(L + 3), kj, ge32, nullptr, nullptr, xkj16, kT, 128, 1, 0, 0);
    // triplet bilinear -> tm32; segment-sum over ji -> aggr
    BilP bp{xkj16, sbfp, wt_bil[l], tm32, kT};
    k_bilinear<<<(kT + 15) / 16, 256, 0, stream>>>(bp);
    k_zero<<<g1d(EH), 256, 0, stream>>>(aggr, EH);
    k_scatter<<<kT, 128, 0, stream>>>(tm32, ji, aggr, kT);
    // m2 = res2(res1(x_ji + aggr))
    k_addsw16<<<g1d(EH), 256, 0, stream>>>(t1, aggr, t2, a16, EH);
    gemm(a16, wt_ir[l][0], P(L + 9),  nullptr, nullptr, nullptr, nullptr, b16, kE, 128, 0, 1, 0);
    gemm(b16, wt_ir[l][1], P(L + 11), nullptr, nullptr, t2, t2, a16, kE, 128, 0, 0, 1);
    gemm(a16, wt_ir[l][2], P(L + 13), nullptr, nullptr, nullptr, nullptr, b16, kE, 128, 0, 1, 0);
    gemm(b16, wt_ir[l][3], P(L + 15), nullptr, nullptr, t2, t2, a16, kE, 128, 0, 0, 1);
    // m = m + swish(lin_out(m2))
    k_cast16<<<g1d(EH), 256, 0, stream>>>(t2, a16, EH, 0);
    gemm(a16, wt_out[l], P(L + 5), nullptr, nullptr, m32, m32, m16, kE, 128, 1, 0, 0);
    run_out_block(l + 1);
  }

  // -------- pooling + conditioned head
  k_zero<<<g1d((size_t)kB * 128), 256, 0, stream>>>(hg, (size_t)kB * 128);
  k_scatter<<<kN, 128, 0, stream>>>(h32, batch, hg, kN);
  k_head<<<kB, 256, 0, stream>>>(hg, ph, te, rf, box, P(9),
                                 P(11), P(10), P(13), P(12),
                                 P(91), P(90), P(93), P(92), P(95), P(94),
                                 (float*)d_out);
}
